// GNN_Nodes_1047972021082
// MI455X (gfx1250) — matrix operand order repaired and checked by reading, not compiled, not run on hardware
//
#include <hip/hip_runtime.h>

typedef __attribute__((ext_vector_type(16))) _Float16 v16h;
typedef __attribute__((ext_vector_type(8)))  _Float16 v8h;
typedef __attribute__((ext_vector_type(8)))  float    v8f;

#define N_NODES 100000
#define N_EDGES 1600000
#define HDIM    128
#define NLAYERS 3
#define NCLASS  40
#define CPAD    48     // classes padded to 3 WMMA tiles
#define KCAT    512    // F + L*H
#define BN_EPS  1e-5f

// ---------------- utility kernels ----------------

__global__ void k_zero(float* __restrict__ p, int n) {
  int i = blockIdx.x * blockDim.x + threadIdx.x;
  if (i < n) p[i] = 0.f;
}

__global__ void k_deg(const int* __restrict__ dst, float* __restrict__ deg) {
  int e = blockIdx.x * blockDim.x + threadIdx.x;
  if (e < N_EDGES) atomicAdd(&deg[dst[e]], 1.0f);
}

__global__ void k_dinv(float* __restrict__ deg) {
  int i = blockIdx.x * blockDim.x + threadIdx.x;
  if (i < N_NODES) deg[i] = rsqrtf(deg[i] + 1.0f);  // self-loop degree
}

// x -> f16 copy (A-side of layer-0 GEMM and head GEMM)
__global__ void k_prep_x(const float* __restrict__ x, _Float16* __restrict__ xh) {
  int i = blockIdx.x * blockDim.x + threadIdx.x;
  if (i < N_NODES * HDIM) xh[i] = (_Float16)x[i];
}

// transpose + f16-convert layer weights: Wt[l][n][k] = W[l][k][n]
__global__ void k_prep_w(const float* __restrict__ W, _Float16* __restrict__ Wt) {
  int i = blockIdx.x * blockDim.x + threadIdx.x;
  if (i >= NLAYERS * HDIM * HDIM) return;
  int l = i >> 14, r = i & 16383, n = r >> 7, k = r & 127;
  Wt[i] = (_Float16)W[(l << 14) + (k << 7) + n];
}

// transpose + pad + f16-convert head weights: Wot[c][k] = W_out[k][c], c<40 else 0
__global__ void k_prep_wout(const float* __restrict__ Wo, _Float16* __restrict__ Wot) {
  int i = blockIdx.x * blockDim.x + threadIdx.x;
  if (i >= CPAD * KCAT) return;
  int c = i >> 9, k = i & 511;
  Wot[i] = (c < NCLASS) ? (_Float16)Wo[k * NCLASS + c] : (_Float16)0.f;
}

// ---------------- WMMA helpers ----------------

// 16x32 f16 A fragment straight from f16 row-major memory (no converts).
// Lane l holds row (l&15); halves 0..15 = K kb..kb+15,
// where kb = ktile + 16*(l>=16)  (contiguous K order, matches B fragment).
__device__ inline v16h load_a_f16(const _Float16* __restrict__ src, int row, int kb) {
  const _Float16* p = src + (long)row * HDIM + kb;
  v8h a0 = *(const v8h*)(p);        // 16B load
  v8h a1 = *(const v8h*)(p + 8);    // 16B load
  return __builtin_shufflevector(a0, a1, 0, 1, 2, 3, 4, 5, 6, 7,
                                         8, 9, 10, 11, 12, 13, 14, 15);
}

// ---------------- layer GEMM: hw = h @ W  (N x 128 @ 128 x 128) ----------------
// block = 256 threads = 8 waves; wave w computes the 16x16 tile at cols w*16,
// rows blockIdx.x*16. K loop = 4 x v_wmma_f32_16x16x32_f16, f32 accumulate.
__global__ void k_gemm_layer(const _Float16* __restrict__ h,
                             const _Float16* __restrict__ Wt,   // [n][k] transposed
                             float* __restrict__ hw) {
  int lane = threadIdx.x & 31;
  int wave = threadIdx.x >> 5;
  int rowtile = blockIdx.x << 4;
  int coltile = wave << 4;
  int lo = lane & 15;
  int hi16 = lane >> 4;
  int lrow = rowtile + lo;
  int bcol = coltile + lo;
  v8f acc = {};
#pragma unroll
  for (int kt = 0; kt < HDIM; kt += 32) {
    v16h a = load_a_f16(h, lrow, kt + (hi16 << 4));
    // B fragment: lane holds col (lane&15), 16 contiguous K starting kt+16*(lane>=16)
    v16h b = *(const v16h*)(Wt + bcol * HDIM + kt + (hi16 << 4));
    acc = __builtin_amdgcn_wmma_f32_16x16x32_f16(false, a, false, b,
                                                 (short)0, acc, false, false);
  }
  int srow = rowtile + (hi16 << 3);
  float* o = hw + (long)srow * HDIM + coltile + lo;
#pragma unroll
  for (int r = 0; r < 8; ++r) o[(long)r * HDIM] = acc[r];  // D: VGPR r -> row srow+r
}

// ---------------- aggregation ----------------

// agg = hw * dinv^2 (self loop) + bias
__global__ void k_init_agg(const float* __restrict__ hw, const float* __restrict__ dinv,
                           const float* __restrict__ bias, float* __restrict__ agg) {
  int i = blockIdx.x * blockDim.x + threadIdx.x;
  if (i >= N_NODES * HDIM) return;
  int n = i >> 7, f = i & 127;
  float di = dinv[n];
  agg[i] = hw[i] * di * di + bias[f];
}

// one wave32 per edge: 32 lanes x float4 = 128 features, fully coalesced
__global__ void k_scatter(const int* __restrict__ src, const int* __restrict__ dst,
                          const float* __restrict__ dinv,
                          const float* __restrict__ hw, float* __restrict__ agg) {
  unsigned tid = blockIdx.x * blockDim.x + threadIdx.x;
  unsigned e = tid >> 5;
  if (e >= N_EDGES) return;
  int q = (tid & 31) << 2;
  int s = src[e], d = dst[e];
  float w = dinv[s] * dinv[d];
  float4 v = *(const float4*)(hw + (long)s * HDIM + q);
  float* o = agg + (long)d * HDIM + q;
  atomicAdd(o + 0, v.x * w);
  atomicAdd(o + 1, v.y * w);
  atomicAdd(o + 2, v.z * w);
  atomicAdd(o + 3, v.w * w);
}

// ---------------- BatchNorm (training mode, biased var) ----------------

__global__ void k_bn_stats(const float* __restrict__ agg, float* __restrict__ stats) {
  __shared__ float ls[HDIM], lss[HDIM];
  int f = threadIdx.x & 127;
  int part = threadIdx.x >> 7;            // 0 or 1
  long base = (long)blockIdx.x * 128;
  float s = 0.f, ss = 0.f;
  for (int r = part; r < 128; r += 2) {
    long row = base + r;
    if (row < N_NODES) {
      float v = agg[row * HDIM + f];
      s += v; ss += v * v;
    }
  }
  if (part == 1) { ls[f] = s; lss[f] = ss; }
  __syncthreads();
  if (part == 0) {
    atomicAdd(&stats[f],        s  + ls[f]);
    atomicAdd(&stats[HDIM + f], ss + lss[f]);
  }
}

// normalize and emit the layer output directly in f16 (A-side of next GEMMs)
__global__ void k_bn_apply(const float* __restrict__ agg, const float* __restrict__ stats,
                           const float* __restrict__ gamma, const float* __restrict__ beta,
                           _Float16* __restrict__ hout) {
  int i = blockIdx.x * blockDim.x + threadIdx.x;
  if (i >= N_NODES * HDIM) return;
  int f = i & 127;
  const float inv_n = 1.0f / (float)N_NODES;
  float mu  = stats[f] * inv_n;
  float var = stats[HDIM + f] * inv_n - mu * mu;
  float v = gamma[f] * (agg[i] - mu) * rsqrtf(var + BN_EPS) + beta[f];
  hout[i] = (_Float16)v;
}

// ---------------- head: out = [x|h1|h2|h3] @ W_out + b_out ----------------
// one wave per 16x16 output tile; K = 512 split over the 4 f16 source buffers.
__global__ void k_out_gemm(const _Float16* __restrict__ x,  const _Float16* __restrict__ h1,
                           const _Float16* __restrict__ h2, const _Float16* __restrict__ h3,
                           const _Float16* __restrict__ Wot, const float* __restrict__ bo,
                           float* __restrict__ out) {
  int lane = threadIdx.x & 31;
  int rowtile = blockIdx.x << 4;
  int coltile = blockIdx.y << 4;
  int lo = lane & 15;
  int hi16 = lane >> 4;
  int lrow = rowtile + lo;
  int bcol = coltile + lo;
  const _Float16* bufs[4] = {x, h1, h2, h3};
  v8f acc = {};
#pragma unroll
  for (int bsel = 0; bsel < 4; ++bsel) {
    const _Float16* __restrict__ src = bufs[bsel];
#pragma unroll
    for (int kk = 0; kk < HDIM; kk += 32) {
      v16h a = load_a_f16(src, lrow, kk + (hi16 << 4));
      int kglob = (bsel << 7) + kk;
      v16h b = *(const v16h*)(Wot + bcol * KCAT + kglob + (hi16 << 4));
      acc = __builtin_amdgcn_wmma_f32_16x16x32_f16(false, a, false, b,
                                                   (short)0, acc, false, false);
    }
  }
  int col = coltile + lo;
  if (col < NCLASS) {
    float bb = bo[col];
    int srow = rowtile + (hi16 << 3);
#pragma unroll
    for (int r = 0; r < 8; ++r) out[(long)(srow + r) * NCLASS + col] = acc[r] + bb;
  }
}

// ---------------- host orchestration ----------------

extern "C" void kernel_launch(void* const* d_in, const int* in_sizes, int n_in,
                              void* d_out, int out_size, void* d_ws, size_t ws_size,
                              hipStream_t stream) {
  const float* x      = (const float*)d_in[0];
  const float* Ws     = (const float*)d_in[1];
  const float* bs     = (const float*)d_in[2];
  const float* gammas = (const float*)d_in[3];
  const float* betas  = (const float*)d_in[4];
  const float* W_out  = (const float*)d_in[5];
  const float* b_out  = (const float*)d_in[6];
  const int*   ei     = (const int*)d_in[7];
  const int* esrc = ei;             // edge_index[0]
  const int* edst = ei + N_EDGES;   // edge_index[1]
  float* out = (float*)d_out;

  // carve workspace (256B aligned slices)
  char* p = (char*)d_ws;
  auto take = [&](size_t bytes) {
    char* r = p; p += (bytes + 255) & ~(size_t)255; return (void*)r;
  };
  float*    dinv  = (float*)take(sizeof(float) * N_NODES);
  float*    hw    = (float*)take(sizeof(float) * (size_t)N_NODES * HDIM);
  float*    agg   = (float*)take(sizeof(float) * (size_t)N_NODES * HDIM);
  _Float16* xh    = (_Float16*)take(sizeof(_Float16) * (size_t)N_NODES * HDIM);
  _Float16* hbuf[NLAYERS];
  for (int i = 0; i < NLAYERS; ++i)
    hbuf[i] = (_Float16*)take(sizeof(_Float16) * (size_t)N_NODES * HDIM);
  _Float16* Wt    = (_Float16*)take(sizeof(_Float16) * NLAYERS * HDIM * HDIM);
  _Float16* Wot   = (_Float16*)take(sizeof(_Float16) * CPAD * KCAT);
  float*    stats = (float*)take(sizeof(float) * 2 * HDIM);

  // degree -> dinv (recomputed each call: no cross-call state)
  k_zero<<<(N_NODES + 255) / 256, 256, 0, stream>>>(dinv, N_NODES);
  k_deg <<<(N_EDGES + 255) / 256, 256, 0, stream>>>(edst, dinv);
  k_dinv<<<(N_NODES + 255) / 256, 256, 0, stream>>>(dinv);

  // f16 operand prep
  k_prep_x   <<<(N_NODES * HDIM + 255) / 256,        256, 0, stream>>>(x, xh);
  k_prep_w   <<<(NLAYERS * HDIM * HDIM + 255) / 256, 256, 0, stream>>>(Ws, Wt);
  k_prep_wout<<<(CPAD * KCAT + 255) / 256,           256, 0, stream>>>(W_out, Wot);

  const _Float16* h = xh;
  for (int l = 0; l < NLAYERS; ++l) {
    k_gemm_layer<<<N_NODES / 16, 256, 0, stream>>>(h, Wt + (size_t)l * HDIM * HDIM, hw);
    k_init_agg<<<(N_NODES * HDIM + 255) / 256, 256, 0, stream>>>(hw, dinv, bs + l * HDIM, agg);
    k_scatter<<<(unsigned)(((size_t)N_EDGES * 32) / 256), 256, 0, stream>>>(esrc, edst, dinv, hw, agg);
    k_zero<<<1, 256, 0, stream>>>(stats, 2 * HDIM);
    k_bn_stats<<<(N_NODES + 127) / 128, 256, 0, stream>>>(agg, stats);
    k_bn_apply<<<(N_NODES * HDIM + 255) / 256, 256, 0, stream>>>(agg, stats,
        gammas + l * HDIM, betas + l * HDIM, hbuf[l]);
    h = hbuf[l];
  }

  dim3 g(N_NODES / 16, (CPAD / 16));
  k_out_gemm<<<g, 32, 0, stream>>>(xh, hbuf[0], hbuf[1], hbuf[2], Wot, b_out, out);
}
